// ManualCPLinear_6150393168456
// MI455X (gfx1250) — matrix operand order repaired and checked by reading, not compiled, not run on hardware
//
#include <hip/hip_runtime.h>
#include <hip/hip_bf16.h>

// ---------------------------------------------------------------------------
// ManualCPLinear on MI455X (gfx1250), wave32 + WMMA.
//
// out[t,m] = bias[m] + sum_r a_q[m,r] * sum_n xq[t,n] * W[r,n]
//   where W[r, i*64+j] = b_q[i,r] * c_q_t[r,j]  and xq = nvfp4 fake-quant.
// Two f16 WMMA GEMMs per 16-token tile; f32 accumulate. Memory-bound
// (268 MB HBM traffic ~ 11.5us at 23.3 TB/s; math ~ 1.7us at f16 WMMA).
// ---------------------------------------------------------------------------

typedef __attribute__((ext_vector_type(16))) _Float16 v16h;
typedef __attribute__((ext_vector_type(8)))  _Float16 v8h;
typedef __attribute__((ext_vector_type(4)))  _Float16 v4h;
typedef __attribute__((ext_vector_type(8)))  float    v8f;

#define NDIM    4096
#define RANK    64
#define TOKENS  8192
#define TILE_T  16
#define CHUNK   512
#define NCHUNKS (NDIM / CHUNK)          // 8
#define KSTEPS  (CHUNK / 32)            // 16 wmma k-steps per chunk
#define XQ_STRIDE (CHUNK + 8)           // +8 halves pad -> conflict-free-ish ds_load_b128
#define H2_STRIDE (RANK + 8)

// E2M1 nearest-grid round (grid 0,.5,1,1.5,2,3,4,6; mids .25,.75,1.25,1.75,2.5,3.5,5)
__device__ __forceinline__ float fp4_round(float y) {
  float a = fabsf(y);
  float q = (a < 0.25f) ? 0.0f :
            (a < 0.75f) ? 0.5f :
            (a < 1.25f) ? 1.0f :
            (a < 1.75f) ? 1.5f :
            (a < 2.5f)  ? 2.0f :
            (a < 3.5f)  ? 3.0f :
            (a < 5.0f)  ? 4.0f : 6.0f;
  return copysignf(q, y);
}

__device__ __forceinline__ v16h cat16(v8h lo, v8h hi) {
  return __builtin_shufflevector(lo, hi, 0,1,2,3,4,5,6,7,8,9,10,11,12,13,14,15);
}

// ---------------------------------------------------------------------------
// Kernel 0: fold b_q/c_q_t into Wt[r][n] (K-major for WMMA B-frags) and
// convert a_q to f16 into At[m][r]. 1 MB f16 total -> L2 resident for main kernel.
// ---------------------------------------------------------------------------
__global__ __launch_bounds__(256) void prep_weights(
    const float* __restrict__ a_q,    // (4096, 64)
    const float* __restrict__ b_q,    // (64, 64)
    const float* __restrict__ c_q_t,  // (64, 64)
    _Float16* __restrict__ Wt,        // (64, 4096)
    _Float16* __restrict__ At)        // (4096, 64)
{
  int i = blockIdx.x * blockDim.x + threadIdx.x;
  if (i < NDIM * RANK) {
    int r = i >> 12, n = i & (NDIM - 1);              // Wt[r][n] = b[n/64,r]*c[r,n%64]
    Wt[i] = (_Float16)(b_q[(n >> 6) * 64 + r] * c_q_t[r * 64 + (n & 63)]);
    int m = i >> 6, r2 = i & 63;                      // At[m][r2] = a_q[m][r2]
    At[i] = (_Float16)(a_q[m * RANK + r2]);
  }
}

// ---------------------------------------------------------------------------
// Main kernel: one 16-token tile per workgroup, 4 waves (128 threads).
// ---------------------------------------------------------------------------
__global__ __launch_bounds__(128) void cplinear_kernel(
    const float* __restrict__ x,       // (8192, 4096)
    const float* __restrict__ smooth,  // (4096)
    const _Float16* __restrict__ Wt,   // (64, 4096)
    const _Float16* __restrict__ At,   // (4096, 64)
    const float* __restrict__ bias,    // (4096)
    float* __restrict__ out)           // (8192, 4096)
{
  __shared__ _Float16 xq[TILE_T][XQ_STRIDE];   // quantized activation chunk (f16)
  __shared__ _Float16 h2s[TILE_T][H2_STRIDE];  // H2 tile staged for GEMM2 A-frags

  const int tid  = threadIdx.x;
  const int lane = tid & 31;
  const int wave = tid >> 5;
  const size_t t0 = (size_t)blockIdx.x * TILE_T;
  const float* xbase = x + t0 * NDIM;

  // ---- GEMM1: H2(16 x 64) = Xq(16 x 4096) @ Wt, wave w owns r-tile w*16 ----
  v8f acc1 = {0.f,0.f,0.f,0.f,0.f,0.f,0.f,0.f};
  const int r0 = wave * 16;
  const int arow = lane & 15;
  const int khi  = (lane >> 4) * 8;    // ISA 16-bit A layout: lanes 16-31 hold K+8

  for (int c = 0; c < NCHUNKS; ++c) {
    const int kbase = c * CHUNK;

    // Prefetch next chunk of x into cache (global_prefetch_b8).
    if (c + 1 < NCHUNKS) {
      const float* p = xbase + (size_t)(tid >> 3) * NDIM + kbase + CHUNK + (tid & 7) * 64;
      __builtin_prefetch(p, 0, 3);
      __builtin_prefetch(p + 32, 0, 3);
    }

    // --- quantize 16 x 512 chunk: iter 'it' handles row 'it', coalesced ---
    for (int it = 0; it < TILE_T; ++it) {
      const int col = tid * 4;                      // 128 thr * 4 = 512 cols
      float4 v = *(const float4*)(xbase + (size_t)it * NDIM + kbase + col);
      float4 s = *(const float4*)(smooth + kbase + col);
      float vx = v.x * s.x, vy = v.y * s.y, vz = v.z * s.z, vw = v.w * s.w;
      // block of 16 = 4 adjacent lanes (cols are 4/lane): amax via shfl_xor
      float am = fmaxf(fmaxf(fabsf(vx), fabsf(vy)), fmaxf(fabsf(vz), fabsf(vw)));
      am = fmaxf(am, __shfl_xor(am, 1, 32));
      am = fmaxf(am, __shfl_xor(am, 2, 32));
      float scale = fmaxf(am * (1.0f / 6.0f), 1e-8f);
      float inv = 1.0f / scale;
      v4h q;
      q[0] = (_Float16)(fp4_round(vx * inv) * scale);
      q[1] = (_Float16)(fp4_round(vy * inv) * scale);
      q[2] = (_Float16)(fp4_round(vz * inv) * scale);
      q[3] = (_Float16)(fp4_round(vw * inv) * scale);
      *(v4h*)&xq[it][col] = q;                       // ds_store_b64
    }
    __syncthreads();

    // --- 16 k-steps of v_wmma_f32_16x16x32_f16 over this chunk ---
    for (int ks = 0; ks < KSTEPS; ++ks) {
      const int k0 = ks * 32;
      // A-frag: lane<16 -> row=lane, K {k0..k0+7, k0+16..k0+23}; lane>=16 -> +8
      v8h alo = *(const v8h*)&xq[arow][k0 + khi];
      v8h ahi = *(const v8h*)&xq[arow][k0 + 16 + khi];
      v16h afrag = cat16(alo, ahi);
      // B-frag: N = r0 + lane%16, K same order as A-frag, Wt[r][n] rows
      const _Float16* brow = Wt + (size_t)(r0 + arow) * NDIM + kbase + k0 + khi;
      v16h bfrag = cat16(*(const v8h*)brow, *(const v8h*)(brow + 16));
      acc1 = __builtin_amdgcn_wmma_f32_16x16x32_f16(
          false, afrag, false, bfrag, (short)0, acc1, false, false);
    }
    __syncthreads();  // xq reused next chunk
  }

  // --- spill H2 to LDS in natural row-major (C/D layout: vgpr g -> row g|g+8) ---
  {
    const int rcol = r0 + (lane & 15);
    const int trow = (lane >> 4) * 8;
#pragma unroll
    for (int g = 0; g < 8; ++g)
      h2s[trow + g][rcol] = (_Float16)acc1[g];
  }
  __syncthreads();

  // ---- GEMM2: Out(16 x 4096) = H2(16 x 64) @ At, + bias. 2 k-steps. ----
  v16h ha0, ha1;
  {
    ha0 = cat16(*(const v8h*)&h2s[arow][khi],
                *(const v8h*)&h2s[arow][16 + khi]);
    ha1 = cat16(*(const v8h*)&h2s[arow][32 + khi],
                *(const v8h*)&h2s[arow][48 + khi]);
  }
  const int ncol = lane & 15;
  const int rhalf = lane >> 4;
  for (int nt = 0; nt < 64; ++nt) {
    const int m0 = nt * 64 + wave * 16;
    const _Float16* b0p = At + (size_t)(m0 + ncol) * RANK + khi;  // K same order as ha0
    const _Float16* b1p = b0p + 32;                               // K + 32
    v16h bf0 = cat16(*(const v8h*)b0p, *(const v8h*)(b0p + 16));
    v16h bf1 = cat16(*(const v8h*)b1p, *(const v8h*)(b1p + 16));
    v8f acc2 = {0.f,0.f,0.f,0.f,0.f,0.f,0.f,0.f};
    acc2 = __builtin_amdgcn_wmma_f32_16x16x32_f16(
        false, ha0, false, bf0, (short)0, acc2, false, false);
    acc2 = __builtin_amdgcn_wmma_f32_16x16x32_f16(
        false, ha1, false, bf1, (short)0, acc2, false, false);
    const int mcol = m0 + ncol;
    const float bv = bias[mcol];
    float* orow = out + (t0 + rhalf * 8) * NDIM + mcol;
#pragma unroll
    for (int g = 0; g < 8; ++g)
      orow[(size_t)g * NDIM] = acc2[g] + bv;
  }
}

// ---------------------------------------------------------------------------
extern "C" void kernel_launch(void* const* d_in, const int* in_sizes, int n_in,
                              void* d_out, int out_size, void* d_ws, size_t ws_size,
                              hipStream_t stream) {
  const float* x       = (const float*)d_in[0];
  const float* smooth  = (const float*)d_in[1];
  const float* a_q     = (const float*)d_in[2];
  const float* b_q     = (const float*)d_in[3];
  const float* c_q_t   = (const float*)d_in[4];
  const float* bias    = (const float*)d_in[5];
  float* out = (float*)d_out;

  _Float16* Wt = (_Float16*)d_ws;                 // 4096*64 f16 = 512 KB
  _Float16* At = Wt + (size_t)NDIM * RANK;        // 64*4096 f16 = 512 KB

  prep_weights<<<(NDIM * RANK + 255) / 256, 256, 0, stream>>>(a_q, b_q, c_q_t, Wt, At);
  cplinear_kernel<<<TOKENS / TILE_T, 128, 0, stream>>>(x, smooth, Wt, At, bias, out);
}
